// GCN_29429115912800
// MI455X (gfx1250) — compile-verified, matrix operand order checked by reading, not run on hardware
//
#include <hip/hip_runtime.h>

typedef _Float16 half16 __attribute__((ext_vector_type(16)));
typedef float    float8 __attribute__((ext_vector_type(8)));

#define IN_DIM  256
#define HID_DIM 128
#define OUT_DIM 64

// ---------------------------------------------------------------------------
// Degree / normalization helpers
// ---------------------------------------------------------------------------
__global__ void deg_init_kernel(float* __restrict__ deg, int n) {
    int i = blockIdx.x * blockDim.x + threadIdx.x;
    if (i < n) deg[i] = 1.0f;               // self-loop weight
}

__global__ void deg_accum_kernel(const long long* __restrict__ col,
                                 const float* __restrict__ w,
                                 float* __restrict__ deg, int e) {
    int i = blockIdx.x * blockDim.x + threadIdx.x;
    if (i < e) atomicAdd(&deg[col[i]], w[i]);
}

__global__ void dinv_kernel(float* __restrict__ deg, int n) {
    int i = blockIdx.x * blockDim.x + threadIdx.x;
    if (i < n) {
        float d = deg[i];
        deg[i] = (d > 0.0f) ? rsqrtf(d) : 0.0f;   // deg buffer becomes dinv
    }
}

// ---------------------------------------------------------------------------
// Weight prepack: W[K,N] f32 (row-major) -> f16 fragment-major layout
//   P[((ks*(N/16) + tileN)*32 + lane)*16 + j]
// where lane r=lane&15 owns column tileN*16+r and j indexes the ISA 16-bit
// B-fragment K striping (j<8 -> K=ks*32+kb+j ; j>=8 -> K=ks*32+16+kb+j-8,
// kb = (lane>>4)*8). One thread per packed element (K*N total).
// ---------------------------------------------------------------------------
template <int K, int N>
__global__ void pack_w_kernel(const float* __restrict__ W, _Float16* __restrict__ P) {
    int idx = blockIdx.x * blockDim.x + threadIdx.x;
    if (idx >= K * N) return;
    const int j    = idx & 15;
    const int lane = (idx >> 4) & 31;
    const int rest = idx >> 9;
    const int tn   = rest % (N / 16);
    const int ks   = rest / (N / 16);
    const int r = lane & 15, kb = (lane >> 4) * 8;
    const int k = ks * 32 + (j < 8 ? kb + j : 16 + kb + (j - 8));
    const int c = tn * 16 + r;
    P[idx] = (_Float16)W[(size_t)k * N + c];
}

// ---------------------------------------------------------------------------
// WMMA GEMM + fused aggregation epilogue:
//   H[M,N]   = op(A) @ B          (op = identity or ReLU, f16 mul / f32 acc)
//   AGG[M,N] = bias + dinv^2 * H  (self-loop + bias init for scatter phase)
// One wave per 16x16 C tile; blockDim.x = 32*(N/16); gridDim.x = ceil(M/16).
// B comes prepacked f16 fragment-major (one 32B vector load per K-step).
// TAIL=false when M % 16 == 0 (no store predication -> no exec juggling).
// ---------------------------------------------------------------------------
template <int K, int N, bool RELU_A, bool TAIL>
__global__ void gemm_wmma_f16_kernel(const float* __restrict__ A,
                                     const _Float16* __restrict__ Bp,
                                     float* __restrict__ H,
                                     const float* __restrict__ dinv,
                                     const float* __restrict__ bias,
                                     float* __restrict__ AGG, int M) {
    const int lane  = threadIdx.x & 31;
    const int wave  = threadIdx.x >> 5;
    const int tileM = blockIdx.x;
    const int tileN = wave;                 // block spans all N tiles
    const int r     = lane & 15;
    const int hsel  = lane >> 4;
    const int kb    = hsel * 8;

    const int rowA = tileM * 16 + r;
    const int rowL = TAIL ? (rowA < M ? rowA : M - 1) : rowA; // EXEC stays all-1s
    const half16* __restrict__ Bv = reinterpret_cast<const half16*>(Bp);

    float8 acc = {};
#pragma unroll
    for (int ks = 0; ks < K / 32; ++ks) {
        const int k0 = ks * 32;
        // ---- A fragment: lane = row; a[0..7]=K(k0+kb..+7), a[8..15]=K(k0+16+kb..+7)
        const float4* pa0 = reinterpret_cast<const float4*>(A + (size_t)rowL * K + k0 + kb);
        const float4* pa1 = reinterpret_cast<const float4*>(A + (size_t)rowL * K + k0 + 16 + kb);
        float4 a00 = pa0[0], a01 = pa0[1];
        float4 a10 = pa1[0], a11 = pa1[1];
        float va[16] = {a00.x, a00.y, a00.z, a00.w, a01.x, a01.y, a01.z, a01.w,
                        a10.x, a10.y, a10.z, a10.w, a11.x, a11.y, a11.z, a11.w};
        half16 a;
#pragma unroll
        for (int j = 0; j < 16; ++j) {
            float v = va[j];
            if (RELU_A) v = fmaxf(v, 0.0f);
            a[j] = (_Float16)v;
        }
        // ---- B fragment: single 32B vector load from prepacked layout
        half16 b = Bv[((size_t)ks * (N / 16) + tileN) * 32 + lane];
        acc = __builtin_amdgcn_wmma_f32_16x16x32_f16(
            /*neg_a=*/false, a, /*neg_b=*/false, b,
            /*c_mod=*/(short)0, acc, /*reuse_a=*/false, /*reuse_b=*/false);
    }
    // ---- fused store: H and AGG = bias + dinv^2 * H
    const int   col  = tileN * 16 + r;
    const float bval = bias[col];
#pragma unroll
    for (int i = 0; i < 8; ++i) {
        const int row = tileM * 16 + hsel * 8 + i;
        if (!TAIL || row < M) {
            const size_t o = (size_t)row * N + col;
            const float v = acc[i];
            H[o] = v;
            const float di = dinv[row];
            AGG[o] = fmaf(di * di, v, bval);
        }
    }
}

// ---------------------------------------------------------------------------
// Edge scatter-add: one wave32 per edge; lane j owns features j, j+32, ...
// Coalesced gathers of h[row,:] (L2-resident) and cacheline-batched f32
// atomics on out[col,:]. Prefetch the gathered row as soon as src is known.
// ---------------------------------------------------------------------------
template <int F>
__global__ void scatter_add_kernel(const long long* __restrict__ row,
                                   const long long* __restrict__ col,
                                   const float* __restrict__ w,
                                   const float* __restrict__ dinv,
                                   const float* __restrict__ h,
                                   float* __restrict__ out, int e) {
    const int wavesPerBlock = blockDim.x >> 5;
    const int eidx = blockIdx.x * wavesPerBlock + (threadIdx.x >> 5);
    const int lane = threadIdx.x & 31;
    if (eidx >= e) return;
    const long long src = row[eidx];
    const float* __restrict__ hs = h + (size_t)src * F;
    __builtin_prefetch(hs + lane, 0, 0);            // global_prefetch_b8
    const long long dst = col[eidx];
    const float norm = dinv[src] * w[eidx] * dinv[dst];
    float* __restrict__ od = out + (size_t)dst * F;
#pragma unroll
    for (int j = 0; j < F / 32; ++j) {
        atomicAdd(&od[lane + j * 32], norm * hs[lane + j * 32]);
    }
}

// ---------------------------------------------------------------------------
static inline int ceil_div(int a, int b) { return (a + b - 1) / b; }

extern "C" void kernel_launch(void* const* d_in, const int* in_sizes, int n_in,
                              void* d_out, int out_size, void* d_ws, size_t ws_size,
                              hipStream_t stream) {
    const float*     x   = (const float*)d_in[0];
    const long long* ei  = (const long long*)d_in[1];   // int64 [2, E]
    const float*     w   = (const float*)d_in[2];
    const float*     W1  = (const float*)d_in[3];
    const float*     b1  = (const float*)d_in[4];
    const float*     W2  = (const float*)d_in[5];
    const float*     b2  = (const float*)d_in[6];
    float*           out = (float*)d_out;

    const int n = in_sizes[0] / IN_DIM;
    const int e = in_sizes[1] / 2;
    const long long* erow = ei;
    const long long* ecol = ei + e;

    // Workspace carve-up (floats), 256-element aligned segments.
    float* ws = (float*)d_ws;
    size_t o = 0;
    auto carve = [&](size_t cnt) { float* p = ws + o; o += (cnt + 255) & ~(size_t)255; return p; };
    float*    dinv = carve((size_t)n);                        // deg then dinv
    float*    h1   = carve((size_t)n * HID_DIM);              // x @ W1
    float*    a1   = carve((size_t)n * HID_DIM);              // aggregated layer-1
    float*    h2   = carve((size_t)n * OUT_DIM);              // relu(a1) @ W2
    _Float16* W1p  = (_Float16*)carve((size_t)IN_DIM * HID_DIM / 2);  // packed f16
    _Float16* W2p  = (_Float16*)carve((size_t)HID_DIM * OUT_DIM / 2); // packed f16

    const int T = 256;

    // 0) prepack weights into f16 fragment-major layout
    pack_w_kernel<IN_DIM, HID_DIM><<<ceil_div(IN_DIM * HID_DIM, T), T, 0, stream>>>(W1, W1p);
    pack_w_kernel<HID_DIM, OUT_DIM><<<ceil_div(HID_DIM * OUT_DIM, T), T, 0, stream>>>(W2, W2p);

    // 1) degrees & symmetric norm factors
    deg_init_kernel<<<ceil_div(n, T), T, 0, stream>>>(dinv, n);
    deg_accum_kernel<<<ceil_div(e, T), T, 0, stream>>>(ecol, w, dinv, e);
    dinv_kernel<<<ceil_div(n, T), T, 0, stream>>>(dinv, n);

    // 2) layer 1: h1 = x @ W1 ; a1 = b1 + dinv^2*h1  (WMMA + fused epilogue)
    if (n % 16 == 0)
        gemm_wmma_f16_kernel<IN_DIM, HID_DIM, false, false>
            <<<n / 16, 32 * (HID_DIM / 16), 0, stream>>>(x, W1p, h1, dinv, b1, a1, n);
    else
        gemm_wmma_f16_kernel<IN_DIM, HID_DIM, false, true>
            <<<ceil_div(n, 16), 32 * (HID_DIM / 16), 0, stream>>>(x, W1p, h1, dinv, b1, a1, n);

    // 3) layer 1 edge aggregation into a1
    scatter_add_kernel<HID_DIM><<<ceil_div(e, T / 32), T, 0, stream>>>(erow, ecol, w, dinv, h1, a1, e);

    // 4) layer 2: h2 = relu(a1) @ W2 ; d_out = b2 + dinv^2*h2
    if (n % 16 == 0)
        gemm_wmma_f16_kernel<HID_DIM, OUT_DIM, true, false>
            <<<n / 16, 32 * (OUT_DIM / 16), 0, stream>>>(a1, W2p, h2, dinv, b2, out, n);
    else
        gemm_wmma_f16_kernel<HID_DIM, OUT_DIM, true, true>
            <<<ceil_div(n, 16), 32 * (OUT_DIM / 16), 0, stream>>>(a1, W2p, h2, dinv, b2, out, n);

    // 5) layer 2 edge aggregation directly into d_out
    scatter_add_kernel<OUT_DIM><<<ceil_div(e, T / 32), T, 0, stream>>>(erow, ecol, w, dinv, h2, out, e);
}
